// HypER_26628797235444
// MI455X (gfx1250) — matrix operand order repaired and checked by reading, not compiled, not run on hardware
//
#include <hip/hip_runtime.h>
#include <hip/hip_bf16.h>
#include <math.h>

#define B_   1024
#define D_   200
#define OC_  32
#define FW_  9
#define WOUT_ 192
#define NE_  100000
#define KPAD 224
#define EPS_ 1e-5f

typedef __attribute__((ext_vector_type(8)))  __bf16 v8bf;
typedef __attribute__((ext_vector_type(16))) __bf16 v16bf;
typedef __attribute__((ext_vector_type(8)))  float  v8f;

// ---------------------------------------------------------------------------
// Kernel 1: convert E_ent (NE x 200 f32) -> bf16, K padded to 224 with zeros
// ---------------------------------------------------------------------------
__global__ void cvt_eent_kernel(const float* __restrict__ E, __bf16* __restrict__ out)
{
    size_t total = (size_t)NE_ * KPAD;
    size_t stride = (size_t)gridDim.x * blockDim.x;
    for (size_t t = (size_t)blockIdx.x * blockDim.x + threadIdx.x; t < total; t += stride) {
        size_t row = t / KPAD;
        int    col = (int)(t % KPAD);
        out[t] = (col < D_) ? (__bf16)E[row * D_ + col] : (__bf16)0.0f;
    }
}

// ---------------------------------------------------------------------------
// Kernel 2: k = E_rel[r] @ fc1_W^T + fc1_b   -> kbuf (B x 288)
// one block per batch element, 288 threads (9 waves)
// ---------------------------------------------------------------------------
__global__ void k_kernel(const int* __restrict__ r, const float* __restrict__ E_rel,
                         const float* __restrict__ fc1_W, const float* __restrict__ fc1_b,
                         float* __restrict__ kbuf)
{
    __shared__ float rel[D_];
    int b  = blockIdx.x;
    int rb = r[b];
    for (int t = threadIdx.x; t < D_; t += blockDim.x)
        rel[t] = E_rel[(size_t)rb * D_ + t];
    __syncthreads();
    int j = threadIdx.x;
    if (j < OC_ * FW_) {
        float acc = fc1_b[j];
        const float* w = fc1_W + (size_t)j * D_;
        #pragma unroll 8
        for (int d = 0; d < D_; ++d) acc += rel[d] * w[d];
        kbuf[(size_t)b * (OC_ * FW_) + j] = acc;
    }
}

// ---------------------------------------------------------------------------
// Kernel 3: fused gather+bn0 -> conv+bn1 -> fc+bn2+relu -> y (bf16, K padded)
// 2 batch elements per block (static LDS 53 KB), 256 threads
// ---------------------------------------------------------------------------
__global__ __launch_bounds__(256) void fused_mid_kernel(
    const int* __restrict__ e1, const float* __restrict__ E_ent,
    const float* __restrict__ kbuf,
    const float* __restrict__ fc_W, const float* __restrict__ fc_b,
    const float* __restrict__ bn0_g, const float* __restrict__ bn0_b,
    const float* __restrict__ bn0_m, const float* __restrict__ bn0_v,
    const float* __restrict__ bn1_g, const float* __restrict__ bn1_b,
    const float* __restrict__ bn1_m, const float* __restrict__ bn1_v,
    const float* __restrict__ bn2_g, const float* __restrict__ bn2_b,
    const float* __restrict__ bn2_m, const float* __restrict__ bn2_v,
    __bf16* __restrict__ ypad)
{
    __shared__ float xs[2][208];        // bn0(entity row), padded
    __shared__ float ks[2][OC_ * FW_];  // generated filters
    __shared__ float hs[2][OC_ * WOUT_];// conv output after bn1

    int tid = threadIdx.x;
    int b0  = blockIdx.x * 2;

    float s0 = bn0_g[0] * rsqrtf(bn0_v[0] + EPS_);
    float t0 = bn0_b[0] - bn0_m[0] * s0;

    for (int t = tid; t < 2 * D_; t += 256) {
        int g = t / D_, d = t % D_;
        int e = e1[b0 + g];
        xs[g][d] = E_ent[(size_t)e * D_ + d] * s0 + t0;
    }
    for (int t = tid; t < 2 * OC_ * FW_; t += 256) {
        int g = t / (OC_ * FW_), j = t % (OC_ * FW_);
        ks[g][j] = kbuf[(size_t)(b0 + g) * (OC_ * FW_) + j];
    }
    __syncthreads();

    for (int t = tid; t < 2 * OC_ * WOUT_; t += 256) {
        int g  = t / (OC_ * WOUT_);
        int o  = t % (OC_ * WOUT_);
        int oc = o / WOUT_;
        int w  = o % WOUT_;
        float sum = 0.0f;
        #pragma unroll
        for (int f = 0; f < FW_; ++f) sum += xs[g][w + f] * ks[g][oc * FW_ + f];
        float sc = bn1_g[oc] * rsqrtf(bn1_v[oc] + EPS_);
        hs[g][o] = (sum - bn1_m[oc]) * sc + bn1_b[oc];
    }
    __syncthreads();

    for (int t = tid; t < 2 * KPAD; t += 256) {
        int g = t / KPAD, d = t % KPAD;
        if (d < D_) {
            float acc = fc_b[d];
            const float* w = fc_W + (size_t)d * (OC_ * WOUT_);
            const float* h = hs[g];
            #pragma unroll 8
            for (int j = 0; j < OC_ * WOUT_; ++j) acc += h[j] * w[j];
            float sc = bn2_g[d] * rsqrtf(bn2_v[d] + EPS_);
            acc = (acc - bn2_m[d]) * sc + bn2_b[d];
            acc = fmaxf(acc, 0.0f);
            ypad[(size_t)(b0 + g) * KPAD + d] = (__bf16)acc;
        } else {
            ypad[(size_t)(b0 + g) * KPAD + d] = (__bf16)0.0f;
        }
    }
}

// ---------------------------------------------------------------------------
// Kernel 4: scores = sigmoid(y @ E_ent^T + bias)  via v_wmma_f32_16x16x32_bf16
// block = 160 threads = 5 waves; each wave owns a 64x16 output strip
// (4 M-tiles x 1 N-tile, 4 accumulator sets) so each B tile is loaded once
// per k-step and reused 4x. grid = (1024/64 = 16, 6250/5 = 1250) exact cover.
// Output stored non-temporally (streamed once; keep bf16 E_ent in L2).
// ---------------------------------------------------------------------------
__global__ __launch_bounds__(160) void scores_wmma_kernel(
    const __bf16* __restrict__ ypad, const __bf16* __restrict__ epad,
    const float* __restrict__ bias, float* __restrict__ out)
{
    int lane = threadIdx.x & 31;
    int wave = threadIdx.x >> 5;
    int m0   = blockIdx.x * 64;
    int n0   = (blockIdx.y * 5 + wave) * 16;
    int half = lane >> 4;      // 0: lanes 0-15, 1: lanes 16-31
    int lm   = lane & 15;

    const __bf16* arow = ypad + (size_t)(m0 + lm) * KPAD;  // A: rows m0+lm (+16,+32,+48)
    const __bf16* brow = epad + (size_t)(n0 + lm) * KPAD;  // B: col n0+lm = E_ent row

    v8f acc[4] = {v8f{}, v8f{}, v8f{}, v8f{}};

    #pragma unroll
    for (int kb = 0; kb < KPAD; kb += 32) {
        // B-matrix 32x16 bf16 layout: lanes<16 hold K={kb..kb+7, kb+16..kb+23},
        // lanes>=16 hold K={kb+8..15, kb+24..31}
        const __bf16* bp = brow + kb + half * 8;
        v8bf b0 = *(const v8bf*)(bp);
        v8bf b1 = *(const v8bf*)(bp + 16);
        v16bf Bv = __builtin_shufflevector(b0, b1, 0,1,2,3,4,5,6,7,8,9,10,11,12,13,14,15);

        #pragma unroll
        for (int mt = 0; mt < 4; ++mt) {
            // A-matrix 16x32 bf16 layout: lanes<16 hold K={kb..kb+7, kb+16..kb+23},
            // lanes>=16 hold K={kb+8..15, kb+24..31}
            const __bf16* ap = arow + (size_t)mt * 16 * KPAD + kb + half * 8;
            v8bf a0 = *(const v8bf*)(ap);
            v8bf a1 = *(const v8bf*)(ap + 16);
            v16bf A = __builtin_shufflevector(a0, a1, 0,1,2,3,4,5,6,7,8,9,10,11,12,13,14,15);
            acc[mt] = __builtin_amdgcn_wmma_f32_16x16x32_bf16(
                /*neg_a=*/false, A, /*neg_b=*/false, Bv,
                /*c_mod=*/(short)0, acc[mt], /*reuse_a=*/false, /*reuse_b=*/false);
        }
    }

    // C/D layout: VGPR j -> (M = mtile_base + 8*half + j, N = n0 + lm)
    int col  = n0 + lm;
    float bv = bias[col];
    #pragma unroll
    for (int mt = 0; mt < 4; ++mt) {
        #pragma unroll
        for (int j = 0; j < 8; ++j) {
            int row = m0 + mt * 16 + half * 8 + j;
            float s = acc[mt][j] + bv;
            float sig = 1.0f / (1.0f + __expf(-s));
            __builtin_nontemporal_store(sig, &out[(size_t)row * NE_ + col]);
        }
    }
}

// ---------------------------------------------------------------------------
extern "C" void kernel_launch(void* const* d_in, const int* in_sizes, int n_in,
                              void* d_out, int out_size, void* d_ws, size_t ws_size,
                              hipStream_t stream)
{
    const int*   e1    = (const int*)  d_in[0];
    const int*   r     = (const int*)  d_in[1];
    const float* E_ent = (const float*)d_in[2];
    const float* E_rel = (const float*)d_in[3];
    const float* fc1_W = (const float*)d_in[4];
    const float* fc1_b = (const float*)d_in[5];
    const float* fc_W  = (const float*)d_in[6];
    const float* fc_b  = (const float*)d_in[7];
    const float* bias  = (const float*)d_in[8];
    const float* bn0_g = (const float*)d_in[9];
    const float* bn0_b = (const float*)d_in[10];
    const float* bn0_m = (const float*)d_in[11];
    const float* bn0_v = (const float*)d_in[12];
    const float* bn1_g = (const float*)d_in[13];
    const float* bn1_b = (const float*)d_in[14];
    const float* bn1_m = (const float*)d_in[15];
    const float* bn1_v = (const float*)d_in[16];
    const float* bn2_g = (const float*)d_in[17];
    const float* bn2_b = (const float*)d_in[18];
    const float* bn2_m = (const float*)d_in[19];
    const float* bn2_v = (const float*)d_in[20];
    float* out = (float*)d_out;

    // Workspace layout (bytes):
    //   [0,                44,800,000)  E_ent bf16, NE x 224
    //   [44,800,000,       45,258,752)  y bf16,     B  x 224
    //   [45,258,752,       46,438,400)  kbuf f32,   B  x 288
    char*   ws   = (char*)d_ws;
    __bf16* epad = (__bf16*)(ws);
    __bf16* ypad = (__bf16*)(ws + (size_t)44800000);
    float*  kbuf = (float*) (ws + (size_t)45258752);

    // 1) E_ent -> bf16 (padded)
    cvt_eent_kernel<<<dim3(43750), dim3(512), 0, stream>>>(E_ent, epad);

    // 2) hypernetwork filters
    k_kernel<<<dim3(B_), dim3(OC_ * FW_), 0, stream>>>(r, E_rel, fc1_W, fc1_b, kbuf);

    // 3) gather+bn0+conv+bn1+fc+bn2+relu -> y bf16 (padded)
    fused_mid_kernel<<<dim3(B_ / 2), dim3(256), 0, stream>>>(
        e1, E_ent, kbuf, fc_W, fc_b,
        bn0_g, bn0_b, bn0_m, bn0_v,
        bn1_g, bn1_b, bn1_m, bn1_v,
        bn2_g, bn2_b, bn2_m, bn2_v, ypad);

    // 4) scores GEMM + bias + sigmoid via bf16 WMMA (64x16 per wave)
    scores_wmma_kernel<<<dim3(B_ / 64, (NE_ / 16) / 5), dim3(160), 0, stream>>>(
        ypad, epad, bias, out);
}
